// cross_adjcent_att_56495999811924
// MI455X (gfx1250) — matrix operand order repaired and checked by reading, not compiled, not run on hardware
//
#include <hip/hip_runtime.h>
#include <hip/hip_bf16.h>
#include <math.h>

// ---------------------------------------------------------------------------
// cross-adjacent attention, MI455X (gfx1250, wave32)
//   affinity[b,h,w,k] = <stem[b,:,h,w], stem-neighbor_k>  (zero-padded)
//   out[b,c,h,w]      = sum_k softmax(affinity)[k] * x-neighbor_k
// One workgroup per (b, h) output row.
//   * affinity dots  : 16x16 Gram matrices on v_wmma_f32_16x16x32_bf16
//   * x row staging  : TENSOR_LOAD_TO_LDS (TDM), LDS padding does the
//                      128 -> 132 DWORD stride conversion in the DMA
//   * stem staging   : VALU (needs f32->bf16 + transpose), float4 reads
// ---------------------------------------------------------------------------

typedef __bf16 bf16_t;
typedef bf16_t v16bf __attribute__((ext_vector_type(16)));
typedef float  v8f   __attribute__((ext_vector_type(8)));
typedef unsigned int v4u __attribute__((ext_vector_type(4)));
typedef int          v8i __attribute__((ext_vector_type(8)));
typedef int          v4i __attribute__((ext_vector_type(4)));

namespace {

constexpr int kB = 8, kC = 64, kH = 128, kW = 128;

// LDS layout ---------------------------------------------------------------
// x rows   : f32  [3][64 c][132]      (w-contiguous, index w+1, halo zeroed)
constexpr int X_WS      = 132;
constexpr int X_ELEMS   = 3 * kC * X_WS;
constexpr int X_BYTES   = X_ELEMS * 4;            // 101376
// stem rows: bf16 [3][130 px][72]     (ch-contiguous per pixel, 144B stride
//                                      -> 16B-aligned WMMA operand chunks)
constexpr int STEM_WS   = 72;
constexpr int STEM_ROW  = 130 * STEM_WS;
constexpr int STEM_ELEMS= 3 * STEM_ROW;
constexpr int STEM_BYTES= STEM_ELEMS * 2;         // 56160
constexpr int ZERO_F4   = (X_BYTES + STEM_BYTES) / 16;
// Gram scratch: f32 [8 tiles][3 rows][256]
constexpr int GRAM_OFF  = X_BYTES + STEM_BYTES;   // 157536
constexpr int GRAM_BYTES= 8 * 3 * 256 * 4;        // 24576
// softmax weights: f32 [128 w][8]
constexpr int AFF_OFF   = GRAM_OFF + GRAM_BYTES;  // 182112
constexpr int SMEM_BYTES= AFF_OFF + 128 * 8 * 4;  // 186208  (< 320KB WGP LDS)

} // namespace

__device__ __forceinline__ unsigned lds_off(void* p) {
  // generic -> LDS address space, then 32-bit LDS byte offset for the D#
  return (unsigned)(unsigned long long)(__attribute__((address_space(3))) char*)p;
}

__global__ __launch_bounds__(128, 1)
void cross_adj_att_kernel(const float* __restrict__ stem,
                          const float* __restrict__ x,
                          float* __restrict__ out) {
  extern __shared__ __align__(16) char smem[];
  float*  x_s    = reinterpret_cast<float*>(smem);
  bf16_t* stem_s = reinterpret_cast<bf16_t*>(smem + X_BYTES);
  float*  gram_s = reinterpret_cast<float*>(smem + GRAM_OFF);
  float*  aff_s  = reinterpret_cast<float*>(smem + AFF_OFF);

  const int tid = threadIdx.x;
  const int bh  = blockIdx.x;
  const int b   = bh >> 7;
  const int h   = bh & 127;

  // ---- zero the staged region (gives correct zero-padded halos / OOB rows)
  {
    float4  z4 = make_float4(0.f, 0.f, 0.f, 0.f);
    float4* z  = reinterpret_cast<float4*>(smem);
    for (int i = tid; i < ZERO_F4; i += 128) z[i] = z4;
  }
  __syncthreads();

  // ---- stage x rows h-1..h+1 via Tensor Data Mover -----------------------
  // 2D tile per row r: dim0 = 128 f32 (w), dim1 = 64 lines (c),
  // line stride 16384 elements; LDS pad 4 DWORDs every 128 DWORDs -> X_WS=132.
#if __has_builtin(__builtin_amdgcn_tensor_load_to_lds)
  constexpr bool kUseTDM = true;
  if (tid < 32) {   // wave 0 issues TDM (TDM ignores EXEC: single-wave issue)
    for (int r = 0; r < 3; ++r) {
      const int hh = h + r - 1;
      if (0 <= hh && hh < kH) {
        const unsigned long long ga =
            (unsigned long long)(uintptr_t)(x + ((b * kC) * kH + hh) * kW);
        const unsigned la = lds_off(x_s + r * kC * X_WS + 1);
        v4u g0;
        g0[0] = 1u;                                   // count=1, user D#
        g0[1] = la;                                   // lds_addr (bytes)
        g0[2] = (unsigned)(ga & 0xffffffffu);         // global_addr[31:0]
        g0[3] = (unsigned)((ga >> 32) & 0x01ffffffu)  // global_addr[56:32]
              | (2u << 30);                           // type=2 ("image")
        v8i g1;
        g1[0] = (int)((2u << 16)        // data_size = 4B
                    | (1u << 20)        // pad_enable
                    | (6u << 22)        // pad_interval: every 128 DWORDs
                    | (3u << 25));      // pad_amount:   4 DWORDs
        g1[1] = (int)(128u << 16);      // tensor_dim0 = 128   [79:48]
        g1[2] = (int)(64u  << 16);      // tensor_dim1 = 64    [111:80]
        g1[3] = (int)(128u << 16);      // tile_dim0   = 128   [127:112]
        g1[4] = 64;                     // tile_dim1 = 64, tile_dim2 = 0
        g1[5] = kH * kW;                // tensor_dim0_stride = 16384
        g1[6] = 0;                      // stride hi / dim1_stride lo
        g1[7] = 0;
        v4i g2 = {0, 0, 0, 0}, g3 = {0, 0, 0, 0};     // 2D tensor: unused
#if __has_include(<hip/amd_detail/amd_gfx1250_TDM.h>)
        v8i g4 = {0, 0, 0, 0, 0, 0, 0, 0};            // therock 6-arg form
        __builtin_amdgcn_tensor_load_to_lds(g0, g1, g2, g3, g4, 0);
#else
        __builtin_amdgcn_tensor_load_to_lds(g0, g1, g2, g3, 0);
#endif
      }
    }
  }
#else
  constexpr bool kUseTDM = false;
#endif

  // ---- stage stem rows (bf16, transposed to [pixel][channel]) ------------
  for (int idx = tid; idx < 3 * kC * (kW / 4); idx += 128) {
    const int w4 = (idx & 31) * 4;
    const int c  = (idx >> 5) & 63;
    const int r  = idx >> 11;          // 0..2  (row h-1+r)
    const int hh = h + r - 1;
    if (0 <= hh && hh < kH) {
      const int g = ((b * kC + c) * kH + hh) * kW + w4;
      __builtin_prefetch(&stem[g + kH * kW], 0, 0);   // global_prefetch_b8
      const float4 vs = *reinterpret_cast<const float4*>(&stem[g]);
      bf16_t* sp = stem_s + r * STEM_ROW + (w4 + 1) * STEM_WS + c;
      sp[0 * STEM_WS] = (bf16_t)vs.x;
      sp[1 * STEM_WS] = (bf16_t)vs.y;
      sp[2 * STEM_WS] = (bf16_t)vs.z;
      sp[3 * STEM_WS] = (bf16_t)vs.w;
      if (!kUseTDM) {   // fallback: manual x staging
        const float4 vx = *reinterpret_cast<const float4*>(&x[g]);
        *reinterpret_cast<float4*>(&x_s[(r * kC + c) * X_WS + (w4 + 1)]) = vx;
      }
    }
  }
#if __has_builtin(__builtin_amdgcn_tensor_load_to_lds)
  if (tid < 32) __builtin_amdgcn_s_wait_tensorcnt(0);  // TDM done before barrier
#endif
  __syncthreads();

  // ---- stage 1: Gram matrices via WMMA -----------------------------------
  // wave wv owns pixel tiles 2*wv and 2*wv+1 (16 px each).
  const int  wv     = tid >> 5;
  const int  lane   = tid & 31;
  const bool hiHalf = lane >= 16;
  const int  il     = lane & 15;

  for (int ts = 0; ts < 2; ++ts) {
    const int tIdx = wv * 2 + ts;
    const int wp   = tIdx * 16 + il + 1;                 // LDS pixel index
    const bf16_t* apix = stem_s + 1 * STEM_ROW + wp * STEM_WS;  // center row
    for (int r = 0; r < 3; ++r) {
      const bf16_t* bpix = stem_s + r * STEM_ROW + wp * STEM_WS;
      v8f acc = {};
      #pragma unroll
      for (int kk = 0; kk < 2; ++kk) {                   // K = 64 = 2 x 32
        // A operand (contiguous k order, same as B):
        //  lanes 0-15 : K = kk*32 + 0..15
        //  lanes16-31 : K = kk*32 + 16..31
        v16bf a, bm;
        const bf16_t* ap = apix + kk * 32 + (hiHalf ? 16 : 0);
        #pragma unroll
        for (int j = 0; j < 8; ++j) { a[j] = ap[j]; a[8 + j] = ap[8 + j]; }
        // B operand (32x16): lanes 0-15 K=kk*32+0..15, lanes16-31 K=+16..31
        const bf16_t* bp = bpix + kk * 32 + (hiHalf ? 16 : 0);
        #pragma unroll
        for (int j = 0; j < 16; ++j) bm[j] = bp[j];
        acc = __builtin_amdgcn_wmma_f32_16x16x32_bf16(
                  false, a, false, bm, (short)0, acc, false, false);
      }
      // C/D layout: VGPR i, lanes0-15 -> (M=i,   N=lane)
      //                      lanes16-31-> (M=8+i, N=lane-16)
      float* g  = gram_s + (tIdx * 3 + r) * 256;
      const int mb = hiHalf ? 8 : 0;
      #pragma unroll
      for (int i = 0; i < 8; ++i) g[(mb + i) * 16 + il] = acc[i];
    }
  }
  // Gram reads below are wave-local (DS in-order within a wave).

  // ---- gather 8 logits per pixel, softmax --------------------------------
  const int w   = wv * 32 + lane;                 // this lane's output pixel
  const int ts2 = lane >> 4;
  const float*  gb  = gram_s + (wv * 2 + ts2) * 3 * 256;
  const bf16_t* ctr = stem_s + 1 * STEM_ROW + (w + 1) * STEM_WS;

  float logit[8];
  int   k = 0;
  for (int r = 0; r < 3; ++r) {
    for (int dj = -1; dj <= 1; ++dj) {
      if (r == 1 && dj == 0) continue;            // center excluded
      const int j = il + dj;
      float v;
      if (0 <= j && j < 16) {
        v = gb[r * 256 + il * 16 + j];            // Gram diagonal entry
      } else {
        // tile-edge: short dot against zero-padded LDS halo
        const bf16_t* nb = stem_s + r * STEM_ROW + (w + 1 + dj) * STEM_WS;
        float s = 0.f;
        #pragma unroll 8
        for (int c = 0; c < kC; ++c) s += (float)ctr[c] * (float)nb[c];
        v = s;
      }
      logit[k++] = v;
    }
  }
  float m = logit[0];
  #pragma unroll
  for (int q = 1; q < 8; ++q) m = fmaxf(m, logit[q]);
  float e[8], s = 0.f;
  #pragma unroll
  for (int q = 0; q < 8; ++q) { e[q] = expf(logit[q] - m); s += e[q]; }
  const float inv = 1.f / s;
  #pragma unroll
  for (int q = 0; q < 8; ++q) aff_s[w * 8 + q] = e[q] * inv;
  __syncthreads();

  // ---- stage 2: 8-tap weighted sum over x, exact f32 ---------------------
  const int wo = tid;                              // thread <-> pixel column
  float a8[8];
  #pragma unroll
  for (int q = 0; q < 8; ++q) a8[q] = aff_s[wo * 8 + q];
  const int wp2 = wo + 1;
  for (int c = 0; c < kC; ++c) {
    const float* r0 = x_s + (0 * kC + c) * X_WS + wp2;
    const float* r1 = x_s + (1 * kC + c) * X_WS + wp2;
    const float* r2 = x_s + (2 * kC + c) * X_WS + wp2;
    const float acc = a8[0] * r0[-1] + a8[1] * r0[0] + a8[2] * r0[1]
                    + a8[3] * r1[-1] + a8[4] * r1[1]
                    + a8[5] * r2[-1] + a8[6] * r2[0] + a8[7] * r2[1];
    out[((b * kC + c) * kH + h) * kW + wo] = acc;
  }
}

extern "C" void kernel_launch(void* const* d_in, const int* in_sizes, int n_in,
                              void* d_out, int out_size, void* d_ws, size_t ws_size,
                              hipStream_t stream) {
  const float* stem = (const float*)d_in[0];   // x_stem
  const float* x    = (const float*)d_in[1];   // x
  float*       out  = (float*)d_out;
  dim3 grid(kB * kH), block(128);
  cross_adj_att_kernel<<<grid, block, SMEM_BYTES, stream>>>(stem, x, out);
}
